// SNNLayer_20203526160780
// MI455X (gfx1250) — matrix operand order repaired and checked by reading, not compiled, not run on hardware
//
#include <hip/hip_runtime.h>

typedef __attribute__((ext_vector_type(16))) _Float16 v16h;
typedef __attribute__((ext_vector_type(8)))  float    v8f;

#define SNN_INPUT   256
#define SNN_HIDDEN  512
#define SNN_OUTPUT  256
#define SNN_TSTEPS  16
#define SNN_THRESH  0.5f
#define SNN_SEQ     1024

// One workgroup = 512 threads = 16 wave32s, owns 16 rows of the flattened batch.
// Wave w owns output columns [16w, 16w+16) and hidden k-range [32w, 32w+32).
__global__ __launch_bounds__(512, 1)
void snn_lif_kernel(const float* __restrict__ x,
                    const float* __restrict__ W1,
                    const float* __restrict__ b1,
                    const float* __restrict__ W2,
                    const float* __restrict__ b2,
                    float* __restrict__ out)
{
    // 32KB shared: reused — first as f32 inp1 staging [16][512], then as
    // packed-f16 spike A-fragments [chunk16][lane32][8 dwords] (16KB used).
    __shared__ __align__(16) unsigned char smem[16 * SNN_HIDDEN * 4];
    float*        inp1_lds  = (float*)smem;
    unsigned int* spike_lds = (unsigned int*)smem;

    const int tid     = threadIdx.x;
    const int lane    = tid & 31;
    const int wave    = tid >> 5;
    const int half    = lane >> 4;   // 0: lanes 0-15, 1: lanes 16-31
    const int l16     = lane & 15;
    const int rowBase = blockIdx.x * 16;

    // ---------------- Preload W2 slab as f16 B-fragments (persistent) -------
    // B is K x N (32 x 16 per chunk). VGPR v: K = (v/4)*16 + half*8 + (v%4)*2
    // (+1 in high half). N = l16. B[k][n] = W2[n][k] (row-major).
    const int   bcol  = wave * 16 + l16;
    const float b2v   = b2[bcol];
    const float* W2r  = W2 + (size_t)bcol * SNN_HIDDEN;
    v16h Bfrag[16];
#pragma unroll
    for (int c = 0; c < 16; ++c) {
        v16h b;
#pragma unroll
        for (int v = 0; v < 8; ++v) {
            int k = c * 32 + (v >> 2) * 16 + half * 8 + (v & 3) * 2;
            b[2 * v]     = (_Float16)W2r[k];
            b[2 * v + 1] = (_Float16)W2r[k + 1];
        }
        Bfrag[c] = b;
    }

    // ---------------- Phase 1: inp1 = x @ W1^T + b1 -------------------------
    // Wave w computes inp1 columns [32w, 32w+32) as two 16x16 tiles, K=256.
    const float* xr = x + (size_t)(rowBase + l16) * SNN_INPUT;
#pragma unroll
    for (int tile = 0; tile < 2; ++tile) {
        const int colBase = wave * 32 + tile * 16;
        const float* w1r  = W1 + (size_t)(colBase + l16) * SNN_INPUT;
        v8f acc = {};
#pragma unroll
        for (int c = 0; c < 8; ++c) {
            // A fragment: 16x32 f16, A[m][k] = x[rowBase+m][k], m = l16.
            // VGPR v: k = 32c + (v/4)*16 + half*8 + (v%4)*2 (+1 in high half).
            v16h a;
#pragma unroll
            for (int v = 0; v < 8; ++v) {
                int k = c * 32 + (v >> 2) * 16 + half * 8 + (v & 3) * 2;
                a[2 * v]     = (_Float16)xr[k];
                a[2 * v + 1] = (_Float16)xr[k + 1];
            }
            // B fragment: B[k][n] = W1[colBase+n][k]
            v16h b;
#pragma unroll
            for (int v = 0; v < 8; ++v) {
                int k = c * 32 + (v >> 2) * 16 + half * 8 + (v & 3) * 2;
                b[2 * v]     = (_Float16)w1r[k];
                b[2 * v + 1] = (_Float16)w1r[k + 1];
            }
            acc = __builtin_amdgcn_wmma_f32_16x16x32_f16(
                false, a, false, b, (short)0, acc, false, false);
        }
        const float bias = b1[colBase + l16];
        // C layout: VGPR r, lane -> (m = r + 8*half, n = l16)
#pragma unroll
        for (int r = 0; r < 8; ++r) {
            int m = r + 8 * half;
            inp1_lds[m * SNN_HIDDEN + colBase + l16] = acc[r] + bias;
        }
    }
    __syncthreads();

    // ---------------- Gather owned inp1 into registers ----------------------
    // Thread (wave, lane) owns mem1 row m=l16, k = 32*wave + (v/4)*16 +
    // half*8 + (v%4)*2 (+1) — exactly the A-fragment dword (chunk=wave,
    // lane, vgpr v) it will publish each timestep.
    float inp1r[16], mem1[16];
#pragma unroll
    for (int v = 0; v < 8; ++v) {
        int k = wave * 32 + (v >> 2) * 16 + half * 8 + (v & 3) * 2;
        inp1r[2 * v]     = inp1_lds[l16 * SNN_HIDDEN + k];
        inp1r[2 * v + 1] = inp1_lds[l16 * SNN_HIDDEN + k + 1];
        mem1[2 * v] = 0.0f;
        mem1[2 * v + 1] = 0.0f;
    }
    __syncthreads();   // smem now reused as spike buffer

    // ---------------- Timestep recurrence (all on-chip) ---------------------
    v8f mem2 = {};
    const int bidx  = rowBase >> 10;     // batch index (rows never straddle)
    const int sbase = rowBase & 1023;    // sequence offset
    unsigned int* myspike = spike_lds + (size_t)(wave * 32 + lane) * 8;

    for (int t = 0; t < SNN_TSTEPS; ++t) {
        // mem1 integrate + fire; publish packed-f16 spikes in A-frag layout.
#pragma unroll
        for (int v = 0; v < 8; ++v) {
            unsigned int dw = 0;
#pragma unroll
            for (int hi = 0; hi < 2; ++hi) {
                int   i  = 2 * v + hi;
                float m1 = mem1[i] + inp1r[i];
                bool  sp = m1 > SNN_THRESH;
                mem1[i]  = sp ? 0.0f : m1;
                dw |= (sp ? 0x3C00u : 0u) << (16 * hi);   // f16 1.0 / 0.0
            }
            myspike[v] = dw;
        }
        __syncthreads();

        // mem2 += spike1 @ W2^T : 16 chained WMMAs over the K=512 chunks.
        v8f acc = mem2;
#pragma unroll
        for (int c = 0; c < 16; ++c) {
            const unsigned int* ap = spike_lds + (size_t)(c * 32 + lane) * 8;
            union { uint4 q[2]; v16h h; } u;
            u.q[0] = *(const uint4*)ap;
            u.q[1] = *(const uint4*)(ap + 4);
            acc = __builtin_amdgcn_wmma_f32_16x16x32_f16(
                false, u.h, false, Bfrag[c], (short)0, acc, false, false);
        }

        // + b2, fire, reset, emit spikes to out[b][t][s][col].
#pragma unroll
        for (int r = 0; r < 8; ++r) {
            float m2 = acc[r] + b2v;
            bool  sp = m2 > SNN_THRESH;
            int   m  = r + 8 * half;
            size_t oidx = (((size_t)bidx * SNN_TSTEPS + t) * SNN_SEQ
                           + (size_t)(sbase + m)) * SNN_OUTPUT + bcol;
            out[oidx] = sp ? 1.0f : 0.0f;
            acc[r]    = sp ? 0.0f : m2;
        }
        mem2 = acc;
        __syncthreads();   // protect spike buffer before next overwrite
    }
}

extern "C" void kernel_launch(void* const* d_in, const int* in_sizes, int n_in,
                              void* d_out, int out_size, void* d_ws, size_t ws_size,
                              hipStream_t stream) {
    (void)in_sizes; (void)n_in; (void)d_ws; (void)ws_size; (void)out_size;
    const float* x  = (const float*)d_in[0];
    const float* W1 = (const float*)d_in[1];
    const float* b1 = (const float*)d_in[2];
    const float* W2 = (const float*)d_in[3];
    const float* b2 = (const float*)d_in[4];
    float* out = (float*)d_out;

    const int N = 8 * 1024;            // flattened rows
    dim3 grid(N / 16), block(512);
    snn_lif_kernel<<<grid, block, 0, stream>>>(x, W1, b1, W2, b2, out);
}
